// LSTMTaggerQuantum_65481071395353
// MI455X (gfx1250) — matrix operand order repaired and checked by reading, not compiled, not run on hardware
//
#include <hip/hip_runtime.h>
#include <hip/hip_bf16.h>

typedef __attribute__((ext_vector_type(16))) __bf16 v16bf;
typedef __attribute__((ext_vector_type(8)))  __bf16 v8bf;
typedef __attribute__((ext_vector_type(4)))  __bf16 v4bf;
typedef __attribute__((ext_vector_type(8)))  float  v8f;
typedef __attribute__((ext_vector_type(4)))  float  v4f;

#define Bsz 128
#define Sq  512
#define Ed  512
#define Hd  512
#define G4H 2048      // 4*H
#define KC  1024      // E + H
#define NQd 8
#define Td  50

// LDS A-tile row stride (elements): 1024 + 8 pad -> rows land on distinct
// bank groups (1032*2/4 = 516 dwords; 516 mod 64 = 4 banks/row shift).
#define ASTRIDE 1032

// Workspace layout (bytes)
#define WS_WC    0                          // bf16 [2048][1024]  = 4 MiB
#define WS_BIAS  (4u*1024u*1024u)           // f32  [2048]        = 8 KiB
#define WS_H     (WS_BIAS + 8192u)          // bf16 [128][512]    = 128 KiB
#define WS_C     (WS_H + 131072u)           // f32  [128][512]    = 256 KiB
#define WS_HX    (WS_C + 262144u)           // f32  [128][512]    = 256 KiB
#define WS_Z     (WS_HX + 262144u)          // f32  [128][2048]   = 1 MiB

__device__ __forceinline__ float sigmoidf(float x) { return 1.0f / (1.0f + expf(-x)); }

__device__ __forceinline__ v8f wmma_bf16(v16bf a, v16bf b, v8f c) {
    return __builtin_amdgcn_wmma_f32_16x16x32_bf16(
        /*neg_a=*/false, a, /*neg_b=*/false, b,
        /*c_mod=*/(short)0, c, /*reuse_a=*/false, /*reuse_b=*/false);
}

// ---------------------------------------------------------------------------
// One-time prep: fused weight matrix Wc = [Wih | Whh] in bf16, bias = bih+bhh,
// zero-init h (bf16) and c (f32).
// ---------------------------------------------------------------------------
__global__ void init_kernel(const float* __restrict__ Wih, const float* __restrict__ Whh,
                            const float* __restrict__ bih, const float* __restrict__ bhh,
                            __bf16* __restrict__ Wc, float* __restrict__ bias,
                            __bf16* __restrict__ h, float* __restrict__ c) {
    int idx = blockIdx.x * blockDim.x + threadIdx.x;
    if (idx < G4H * KC) {
        int n = idx / KC, k = idx % KC;
        float w = (k < Ed) ? Wih[n * Ed + k] : Whh[n * Hd + (k - Ed)];
        Wc[idx] = (__bf16)w;
    }
    if (idx < G4H) bias[idx] = bih[idx] + bhh[idx];
    if (idx < Bsz * Hd) { h[idx] = (__bf16)0.0f; c[idx] = 0.0f; }
}

// ---------------------------------------------------------------------------
// Per-timestep GEMM: z[m][n] = sum_k A[m][k] * Wc[n][k] + bias[n]
//   A[m][0:512]    = x[m, t, :]   (f32 -> bf16 during LDS staging)
//   A[m][512:1024] = h_prev[m, :] (bf16 copy during LDS staging)
// Block = 256 threads (8 waves). All waves share one M-tile, so the A tile
// (16 x 1024 bf16, ~33 KB) is staged into LDS once per block; the K-loop then
// reads A via ds_load_b128 and streams B (L2-resident Wc) via global_load_b128.
// Each wave owns one 16x32 output block (2 N-tiles): 2 WMMAs per K-step.
// grid = (8, 8): x = N-pair group / 8, y = M-tile.
// ---------------------------------------------------------------------------
__global__ void lstm_gemm_step(const float* __restrict__ x, const __bf16* __restrict__ Wc,
                               const float* __restrict__ bias, const __bf16* __restrict__ hprev,
                               float* __restrict__ z, int t) {
    __shared__ __align__(16) __bf16 Alds[16 * ASTRIDE];

    const int tid  = threadIdx.x;
    const int lane = tid & 31;
    const int wid  = tid >> 5;
    const int half = lane >> 4;      // 0 or 1
    const int r    = lane & 15;

    const int ngrp  = blockIdx.x * 8 + wid;  // 0..63 (pair of N-tiles)
    const int mtile = blockIdx.y;            // 0..7
    const int n0 = ngrp * 32;

    // ---- Stage A tile into LDS (cooperative, coalesced) ----
    // x half: 16 rows x 512 f32 -> bf16. 8192 elements, 4 per thread per pass.
    {
        const float* xbase = x + (size_t)(mtile * 16) * (Sq * Ed) + (size_t)t * Ed;
#pragma unroll
        for (int p = 0; p < 8; ++p) {
            int flat = (p * 256 + tid) * 4;          // element offset, 0..8191
            int row = flat >> 9, k = flat & 511;
            v4f f = *(const v4f*)(xbase + (size_t)row * (Sq * Ed) + k);
            v4bf bfv;
#pragma unroll
            for (int e = 0; e < 4; ++e) bfv[e] = (__bf16)f[e];
            *(v4bf*)(&Alds[row * ASTRIDE + k]) = bfv;
        }
    }
    // h half: 16 rows x 512 bf16 copy. 8192 elements, 8 per thread per pass.
    {
        const __bf16* hbase = hprev + (size_t)(mtile * 16) * Hd;
#pragma unroll
        for (int p = 0; p < 4; ++p) {
            int flat = (p * 256 + tid) * 8;          // element offset, 0..8191
            int row = flat >> 9, k = flat & 511;
            v8bf hv = *(const v8bf*)(hbase + (size_t)row * Hd + k);
            *(v8bf*)(&Alds[row * ASTRIDE + 512 + k]) = hv;
        }
    }
    __syncthreads();

    const __bf16* w0 = Wc + (size_t)(n0 + r)      * KC;   // N-tile 0 row
    const __bf16* w1 = Wc + (size_t)(n0 + 16 + r) * KC;   // N-tile 1 row
    const __bf16* arow = &Alds[r * ASTRIDE];

    v8f acc0 = {}, acc1 = {};

    // ---- Single branch-free K loop over [0, 1024) ----
    for (int k0 = 0; k0 < KC; k0 += 32) {
        // A fragment from LDS (contiguous k order, same as B):
        //   elements 0..7  <-> K = k0 + half*16 + e
        //   elements 8..15 <-> K = k0 + half*16 + 8 + e
        v8bf lo = *(const v8bf*)(arow + k0 + half * 16);
        v8bf hi = *(const v8bf*)(arow + k0 + half * 16 + 8);
        v16bf a;
#pragma unroll
        for (int e = 0; e < 8; ++e) {
            a[e]     = lo[e];
            a[e + 8] = hi[e];
        }
        // B fragments: lane n, 16 contiguous K values (32B, aligned)
        v16bf b0 = *(const v16bf*)(w0 + k0 + half * 16);
        v16bf b1 = *(const v16bf*)(w1 + k0 + half * 16);
        acc0 = wmma_bf16(a, b0, acc0);
        acc1 = wmma_bf16(a, b1, acc1);
    }

    // C/D layout: lanes 0-15 -> M = v, lanes 16-31 -> M = 8 + v; N = lane&15
    const float bn0 = bias[n0 + r];
    const float bn1 = bias[n0 + 16 + r];
#pragma unroll
    for (int v = 0; v < 8; ++v) {
        const int mrow = mtile * 16 + half * 8 + v;
        z[(size_t)mrow * G4H + n0 + r]      = acc0[v] + bn0;
        z[(size_t)mrow * G4H + n0 + 16 + r] = acc1[v] + bn1;
    }
}

// ---------------------------------------------------------------------------
// Cell update, PyTorch gate order [i, f, g, o]:
//   c = sig(f)*c + sig(i)*tanh(g);  h = sig(o)*tanh(c)
// ---------------------------------------------------------------------------
__global__ void lstm_cell_step(const float* __restrict__ z, float* __restrict__ c,
                               __bf16* __restrict__ h, float* __restrict__ hx, int last) {
    int idx = blockIdx.x * blockDim.x + threadIdx.x;  // 0..65535
    int b = idx >> 9, j = idx & 511;
    const float* zr = z + (size_t)b * G4H;
    float zi = zr[j];
    float zf = zr[512  + j];
    float zg = zr[1024 + j];
    float zo = zr[1536 + j];
    float cn = sigmoidf(zf) * c[idx] + sigmoidf(zi) * tanhf(zg);
    float hn = sigmoidf(zo) * tanhf(cn);
    c[idx] = cn;
    h[idx] = (__bf16)hn;
    if (last) hx[idx] = hn;
}

// ---------------------------------------------------------------------------
// Head: quantum-layer closed form + hidden2tag + log_softmax.
// One block (256 threads) per batch row. The 'i' gate is dead in the reference.
// ---------------------------------------------------------------------------
__global__ void head_kernel(const float* __restrict__ hx,
                            const float* __restrict__ Wf, const float* __restrict__ bf,
                            const float* __restrict__ Wu, const float* __restrict__ bu,
                            const float* __restrict__ Wo, const float* __restrict__ bo,
                            const float* __restrict__ thf, const float* __restrict__ thu,
                            const float* __restrict__ tho,
                            const float* __restrict__ Wt, const float* __restrict__ bt,
                            float* __restrict__ out) {
    __shared__ float part[256];
    __shared__ float dot[24];
    __shared__ float hout[NQd];
    __shared__ float logits[Td];
    __shared__ float red[2];

    const int b = blockIdx.x, tid = threadIdx.x;
    const float* hr = hx + (size_t)b * Hd;

    // 24 dot products (3 gates x 8 wires), 8 threads per dot
    int d = tid >> 3, j = tid & 7;
    float ps = 0.0f;
    if (d < 24) {
        int gate = d >> 3, q = d & 7;
        const float* W = (gate == 0) ? Wf : (gate == 1) ? Wu : Wo;
        const float* wq = W + q * Hd;
        for (int k = j; k < Hd; k += 8) ps += hr[k] * wq[k];
    }
    part[tid] = ps;
    __syncthreads();
    if (tid < 24) {
        float s = 0.0f;
#pragma unroll
        for (int e = 0; e < 8; ++e) s += part[tid * 8 + e];
        dot[tid] = s;
    }
    __syncthreads();
    if (tid < NQd) {
        int q = tid;
        float fv = sigmoidf(cosf(dot[q]      + bf[q] + thf[q]));
        float gv = tanhf  (cosf(dot[8  + q] + bu[q] + thu[q]));
        float ov = sigmoidf(cosf(dot[16 + q] + bo[q] + tho[q]));
        hout[q] = ov * tanhf(fv * gv);
    }
    __syncthreads();
    if (tid < Td) {
        float s = bt[tid];
#pragma unroll
        for (int q = 0; q < NQd; ++q) s += hout[q] * Wt[tid * NQd + q];
        logits[tid] = s;
    }
    __syncthreads();
    if (tid == 0) {
        float m = logits[0];
        for (int t2 = 1; t2 < Td; ++t2) m = fmaxf(m, logits[t2]);
        float se = 0.0f;
        for (int t2 = 0; t2 < Td; ++t2) se += expf(logits[t2] - m);
        red[0] = m;
        red[1] = logf(se);
    }
    __syncthreads();
    if (tid < Td) out[(size_t)b * Td + tid] = logits[tid] - red[0] - red[1];
}

// ---------------------------------------------------------------------------
extern "C" void kernel_launch(void* const* d_in, const int* in_sizes, int n_in,
                              void* d_out, int out_size, void* d_ws, size_t ws_size,
                              hipStream_t stream) {
    (void)in_sizes; (void)n_in; (void)out_size; (void)ws_size;

    const float* x   = (const float*)d_in[0];
    const float* Wih = (const float*)d_in[1];
    const float* Whh = (const float*)d_in[2];
    const float* bih = (const float*)d_in[3];
    const float* bhh = (const float*)d_in[4];
    const float* Wf  = (const float*)d_in[5];
    const float* bf  = (const float*)d_in[6];
    // d_in[7], d_in[8] (Wi, bi) and d_in[14] (thi): dead in the reference
    const float* Wu  = (const float*)d_in[9];
    const float* bu  = (const float*)d_in[10];
    const float* Wo  = (const float*)d_in[11];
    const float* bo  = (const float*)d_in[12];
    const float* thf = (const float*)d_in[13];
    const float* thu = (const float*)d_in[15];
    const float* tho = (const float*)d_in[16];
    const float* Wt  = (const float*)d_in[17];
    const float* bt  = (const float*)d_in[18];

    char* ws = (char*)d_ws;
    __bf16* Wc   = (__bf16*)(ws + WS_WC);
    float*  bias = (float*) (ws + WS_BIAS);
    __bf16* h    = (__bf16*)(ws + WS_H);
    float*  c    = (float*) (ws + WS_C);
    float*  hx   = (float*) (ws + WS_HX);
    float*  z    = (float*) (ws + WS_Z);

    init_kernel<<<(G4H * KC + 255) / 256, 256, 0, stream>>>(Wih, Whh, bih, bhh, Wc, bias, h, c);

    for (int t = 0; t < Sq; ++t) {
        lstm_gemm_step<<<dim3(8, 8), 256, 0, stream>>>(x, Wc, bias, h, z, t);
        lstm_cell_step<<<(Bsz * Hd) / 256, 256, 0, stream>>>(z, c, h, hx, t == Sq - 1);
    }

    head_kernel<<<Bsz, 256, 0, stream>>>(hx, Wf, bf, Wu, bu, Wo, bo, thf, thu, tho, Wt, bt,
                                         (float*)d_out);
}
